// SelectiveSSM_18846316495014
// MI455X (gfx1250) — matrix operand order repaired and checked by reading, not compiled, not run on hardware
//
#include <hip/hip_runtime.h>
#include <hip/hip_bf16.h>

// ---------------------------------------------------------------------------
// Selective SSM (Mamba-style) forward for MI455X / gfx1250 (wave32, WMMA).
//   B=8, T=2048, D=768, N=16, Kconv=4
// Pipeline:
//   k_gemm_xssm : x_ssm = x @ W_in[:D].T       (f16 WMMA, f32 accum, 32x32/wave)
//   k_conv_silu : depthwise causal conv + bias + SiLU
//   k_gemm_dtbc : dt = softplus(x_conv@W_dt.T + b), Bm/Cm = x_conv@W_{B,C}.T
//   k_scan      : sequential selective scan; B_t/C_t staged to LDS with
//                 global_load_async_to_lds_b64 (ASYNCcnt) in 64-step chunks
// ---------------------------------------------------------------------------

#define B_SZ 8
#define T_SZ 2048
#define D_MODEL 768
#define D_STATE 16
#define D_CONV 4
#define LOG2E 1.4426950408889634f
#define LN2   0.6931471805599453f

typedef __attribute__((ext_vector_type(16))) _Float16 v16h;
typedef __attribute__((ext_vector_type(8)))  float    v8f;

#if __has_builtin(__builtin_amdgcn_exp2f)
#define EXP2F(x) __builtin_amdgcn_exp2f(x)
#else
#define EXP2F(x) exp2f(x)
#endif
#if __has_builtin(__builtin_amdgcn_logf)
#define LOG2F(x) __builtin_amdgcn_logf(x)   // base-2 log
#else
#define LOG2F(x) log2f(x)
#endif

__device__ __forceinline__ float silu_f(float v) {
    return v / (1.0f + EXP2F(-v * LOG2E));
}
__device__ __forceinline__ float softplus_f(float v) {
    float e = EXP2F(-fabsf(v) * LOG2E);             // exp(-|v|)
    return fmaxf(v, 0.0f) + LOG2F(1.0f + e) * LN2;  // max(v,0)+log1p(exp(-|v|))
}

// ---- WMMA fragment helpers (A/B in contiguous k order (k = 16h + s), f32 -> f16 on the fly) ----
// A 16x32 f16: lane = row (lane&15); lanes 0-15 hold K 0-15,
// lanes 16-31 hold K 16-31; two f16 packed per VGPR.
__device__ __forceinline__ v16h load_a_frag(const float* __restrict__ arow,
                                            int hi) {
    v16h f;
#pragma unroll
    for (int j = 0; j < 8; ++j) {
        int kb = 16 * hi + 2 * j;
        f[2 * j]     = (_Float16)arow[kb];
        f[2 * j + 1] = (_Float16)arow[kb + 1];
    }
    return f;
}
// B 32x16 f16: lane = column (lane&15); VGPR j holds K pair (2j,2j+1) for
// lanes 0-15 and (16+2j,17+2j) for lanes 16-31.
__device__ __forceinline__ v16h load_b_frag(const float* __restrict__ wrow,
                                            int hi) {
    v16h f;
#pragma unroll
    for (int j = 0; j < 8; ++j) {
        int kb = 16 * hi + 2 * j;
        f[2 * j]     = (_Float16)wrow[kb];
        f[2 * j + 1] = (_Float16)wrow[kb + 1];
    }
    return f;
}
__device__ __forceinline__ v8f wmma16(v16h a, v16h b, v8f c) {
    return __builtin_amdgcn_wmma_f32_16x16x32_f16(
        /*neg_a=*/false, a, /*neg_b=*/false, b,
        /*c_mod=*/(short)0, c, /*reuse_a=*/false, /*reuse_b=*/false);
}
// C/D v8f: lane&15 = N, VGPR i -> M = i + 8*(lane>=16).
__device__ __forceinline__ void store_tile(float* __restrict__ C, size_t ldc,
                                           int m0, int n, v8f a, int hi) {
#pragma unroll
    for (int i = 0; i < 8; ++i)
        C[(size_t)(m0 + i + 8 * hi) * ldc + n] = a[i];
}

// ---- kernel 1: x_ssm = x @ W_in[:D].T  (M=16384, N=768, K=768) ------------
// 8 waves/block arranged 4x2 -> 128x64 block tile; 32x32 per wave (4 WMMAs
// per k-step on 2 A-frags x 2 B-frags).
__global__ __launch_bounds__(256) void k_gemm_xssm(
    const float* __restrict__ x, const float* __restrict__ Win,
    float* __restrict__ xssm) {
    const int wave = threadIdx.x >> 5;
    const int lane = threadIdx.x & 31;
    const int r = lane & 15, hi = lane >> 4, c = r;
    const int m0 = blockIdx.y * 128 + (wave & 3) * 32;
    const int n0 = blockIdx.x * 64 + (wave >> 2) * 32;

    const float* a0 = x + (size_t)(m0 + r) * D_MODEL;
    const float* a1 = a0 + (size_t)16 * D_MODEL;
    const float* w0 = Win + (size_t)(n0 + r) * D_MODEL;
    const float* w1 = w0 + (size_t)16 * D_MODEL;

    v8f acc00 = {}, acc01 = {}, acc10 = {}, acc11 = {};
    for (int k0 = 0; k0 < D_MODEL; k0 += 32) {
        v16h A0 = load_a_frag(a0 + k0, hi);
        v16h A1 = load_a_frag(a1 + k0, hi);
        v16h B0 = load_b_frag(w0 + k0, hi);
        v16h B1 = load_b_frag(w1 + k0, hi);
        acc00 = wmma16(A0, B0, acc00);
        acc01 = wmma16(A0, B1, acc01);
        acc10 = wmma16(A1, B0, acc10);
        acc11 = wmma16(A1, B1, acc11);
    }
    store_tile(xssm, D_MODEL, m0,      n0 + c,      acc00, hi);
    store_tile(xssm, D_MODEL, m0,      n0 + 16 + c, acc01, hi);
    store_tile(xssm, D_MODEL, m0 + 16, n0 + c,      acc10, hi);
    store_tile(xssm, D_MODEL, m0 + 16, n0 + 16 + c, acc11, hi);
}

// ---- kernel 2: depthwise causal conv(K=4) + bias + SiLU -------------------
__global__ __launch_bounds__(256) void k_conv_silu(
    const float* __restrict__ xssm, const float* __restrict__ cw,
    const float* __restrict__ cb, float* __restrict__ xconv) {
    const int idx = blockIdx.x * blockDim.x + threadIdx.x;
    const int total = B_SZ * T_SZ * D_MODEL;
    if (idx >= total) return;
    const int d = idx % D_MODEL;
    const int t = (idx / D_MODEL) % T_SZ;
    const int b = idx / (D_MODEL * T_SZ);
    float acc = cb[d];
#pragma unroll
    for (int k = 0; k < D_CONV; ++k) {
        int ts = t - (D_CONV - 1) + k;
        if (ts >= 0)
            acc = fmaf(cw[d * D_CONV + k],
                       xssm[((size_t)b * T_SZ + ts) * D_MODEL + d], acc);
    }
    xconv[idx] = silu_f(acc);
}

// ---- kernel 3: fused dt / Bm / Cm GEMM ------------------------------------
// blockIdx.x 0..11 : dt 64-wide strips (bias + softplus epilogue)
// blockIdx.x == 12 : Bm (w0=W_B) and Cm (w1=W_C) in one 32x32 wave tile
__global__ __launch_bounds__(256) void k_gemm_dtbc(
    const float* __restrict__ xc, const float* __restrict__ Wdt,
    const float* __restrict__ bdt, const float* __restrict__ WB,
    const float* __restrict__ WC, float* __restrict__ dt,
    float* __restrict__ Bm, float* __restrict__ Cm) {
    const int wave = threadIdx.x >> 5;
    const int lane = threadIdx.x & 31;
    const int r = lane & 15, hi = lane >> 4, c = r;
    const int m0 = blockIdx.y * 128 + (wave & 3) * 32;
    const int wn = wave >> 2;
    const int tn = blockIdx.x;
    const bool special = (tn == 12);
    if (special && wn) return;  // wave-uniform early exit

    int n0 = 0;
    const float *w0, *w1;
    if (!special) {
        n0 = tn * 64 + wn * 32;
        w0 = Wdt + (size_t)(n0 + r) * D_MODEL;
        w1 = w0 + (size_t)16 * D_MODEL;
    } else {
        w0 = WB + (size_t)r * D_MODEL;  // -> Bm
        w1 = WC + (size_t)r * D_MODEL;  // -> Cm
    }
    const float* a0 = xc + (size_t)(m0 + r) * D_MODEL;
    const float* a1 = a0 + (size_t)16 * D_MODEL;

    v8f acc00 = {}, acc01 = {}, acc10 = {}, acc11 = {};
    for (int k0 = 0; k0 < D_MODEL; k0 += 32) {
        v16h A0 = load_a_frag(a0 + k0, hi);
        v16h A1 = load_a_frag(a1 + k0, hi);
        v16h B0 = load_b_frag(w0 + k0, hi);
        v16h B1 = load_b_frag(w1 + k0, hi);
        acc00 = wmma16(A0, B0, acc00);
        acc01 = wmma16(A0, B1, acc01);
        acc10 = wmma16(A1, B0, acc10);
        acc11 = wmma16(A1, B1, acc11);
    }

    if (!special) {
        const float bv0 = bdt[n0 + c];
        const float bv1 = bdt[n0 + 16 + c];
#pragma unroll
        for (int i = 0; i < 8; ++i) {
            dt[(size_t)(m0 + i + 8 * hi) * D_MODEL + n0 + c]       = softplus_f(acc00[i] + bv0);
            dt[(size_t)(m0 + i + 8 * hi) * D_MODEL + n0 + 16 + c]  = softplus_f(acc01[i] + bv1);
            dt[(size_t)(m0 + 16 + i + 8 * hi) * D_MODEL + n0 + c]      = softplus_f(acc10[i] + bv0);
            dt[(size_t)(m0 + 16 + i + 8 * hi) * D_MODEL + n0 + 16 + c] = softplus_f(acc11[i] + bv1);
        }
    } else {
        store_tile(Bm, D_STATE, m0,      c, acc00, hi);
        store_tile(Cm, D_STATE, m0,      c, acc01, hi);
        store_tile(Bm, D_STATE, m0 + 16, c, acc10, hi);
        store_tile(Cm, D_STATE, m0 + 16, c, acc11, hi);
    }
}

// ---- async global -> LDS copy (8B per lane, tracked by ASYNCcnt) ----------
// GVS mode: lds[vdst] = mem[saddr + vaddr].  LDS byte address = low 32 bits
// of the generic pointer.
__device__ __forceinline__ void async_copy_b64(const float* __restrict__ gbase,
                                               unsigned byte_off,
                                               float* lds_dst) {
    unsigned lds_addr = (unsigned)(size_t)lds_dst;
    asm volatile("global_load_async_to_lds_b64 %0, %1, %2"
                 :
                 : "v"(lds_addr), "v"(byte_off), "s"(gbase)
                 : "memory");
}
__device__ __forceinline__ void wait_asynccnt0() {
    asm volatile("s_wait_asynccnt 0" ::: "memory");
}

// ---- kernel 4: selective scan ---------------------------------------------
// One thread per (b,d) channel; h[16] in registers; B_t/C_t staged in LDS in
// 64-step chunks via async global->LDS loads (shared by all 768 d-channels).
#define SCAN_CHUNK 64
__global__ __launch_bounds__(256) void k_scan(
    const float* __restrict__ x, const float* __restrict__ dt,
    const float* __restrict__ Bm, const float* __restrict__ Cm,
    const float* __restrict__ A_log, const float* __restrict__ Dp,
    float* __restrict__ y) {
    __shared__ float sB[SCAN_CHUNK * D_STATE];
    __shared__ float sC[SCAN_CHUNK * D_STATE];
    const int d = blockIdx.x * blockDim.x + threadIdx.x;  // 0..767
    const int b = blockIdx.y;

    float Ad[D_STATE];  // -exp(A_log) * log2(e), pre-folded for exp2
#pragma unroll
    for (int n = 0; n < D_STATE; ++n)
        Ad[n] = -EXP2F(A_log[(size_t)d * D_STATE + n] * LOG2E) * LOG2E;
    const float Dv = Dp[d];

    float h[D_STATE];
#pragma unroll
    for (int n = 0; n < D_STATE; ++n) h[n] = 0.0f;

    for (int t0 = 0; t0 < T_SZ; t0 += SCAN_CHUNK) {
        __syncthreads();  // previous chunk fully consumed
        // stage CHUNK*16 floats of B and C: 512 x 8B each, 256 threads x 2
        const unsigned chunk_byte = ((unsigned)b * T_SZ + t0) * D_STATE * 4u;
#pragma unroll
        for (int j = 0; j < (SCAN_CHUNK * D_STATE / 2) / 256; ++j) {
            const int i2 = threadIdx.x + j * 256;  // float2 index
            async_copy_b64(Bm, chunk_byte + 8u * i2, &sB[2 * i2]);
            async_copy_b64(Cm, chunk_byte + 8u * i2, &sC[2 * i2]);
        }
        wait_asynccnt0();
        __syncthreads();

        for (int tt = 0; tt < SCAN_CHUNK; ++tt) {
            const size_t base = ((size_t)b * T_SZ + (t0 + tt)) * D_MODEL + d;
            const float dtv = dt[base];
            const float xv  = x[base];
            if (tt + 1 < SCAN_CHUNK) {  // global_prefetch_b8 next timestep
                __builtin_prefetch(dt + base + D_MODEL, 0, 1);
                __builtin_prefetch(x + base + D_MODEL, 0, 1);
            }
            const float xdt = xv * dtv;
            float yv = 0.0f;
#pragma unroll
            for (int n = 0; n < D_STATE; ++n) {
                float dA = EXP2F(Ad[n] * dtv);  // exp(A*dt)
                h[n] = fmaf(h[n], dA, xdt * sB[tt * D_STATE + n]);
                yv = fmaf(h[n], sC[tt * D_STATE + n], yv);
            }
            y[base] = fmaf(xv, Dv, yv);
        }
    }
}

// ---------------------------------------------------------------------------
extern "C" void kernel_launch(void* const* d_in, const int* in_sizes, int n_in,
                              void* d_out, int out_size, void* d_ws,
                              size_t ws_size, hipStream_t stream) {
    const float* x      = (const float*)d_in[0];
    const float* W_in   = (const float*)d_in[1];
    const float* conv_w = (const float*)d_in[2];
    const float* conv_b = (const float*)d_in[3];
    const float* W_dt   = (const float*)d_in[4];
    const float* b_dt   = (const float*)d_in[5];
    const float* A_log  = (const float*)d_in[6];
    const float* D_par  = (const float*)d_in[7];
    const float* W_B    = (const float*)d_in[8];
    const float* W_C    = (const float*)d_in[9];
    float* out = (float*)d_out;

    const size_t MT   = (size_t)B_SZ * T_SZ;          // 16384 rows
    const size_t big  = MT * D_MODEL * sizeof(float); // 48 MiB
    const size_t tiny = MT * D_STATE * sizeof(float); // 1 MiB
    char* ws = (char*)d_ws;
    float* xssm  = (float*)(ws);                  // [MT, D]
    float* xconv = (float*)(ws + big);            // [MT, D]
    float* Bmat  = (float*)(ws + 2 * big);        // [MT, N]
    float* Cmat  = (float*)(ws + 2 * big + tiny); // [MT, N]
    float* dtb   = xssm;  // x_ssm dead after conv; reuse for dt

    const dim3 blk(256);
    // 128x64 block tiles: grid.x = N/64, grid.y = M/128
    k_gemm_xssm<<<dim3(D_MODEL / 64, MT / 128), blk, 0, stream>>>(x, W_in,
                                                                  xssm);

    const int total = B_SZ * T_SZ * D_MODEL;
    k_conv_silu<<<dim3((total + 255) / 256), blk, 0, stream>>>(xssm, conv_w,
                                                               conv_b, xconv);

    // 12 dt strips (64-wide) + 1 combined Bm/Cm strip
    k_gemm_dtbc<<<dim3(13, MT / 128), blk, 0, stream>>>(
        xconv, W_dt, b_dt, W_B, W_C, dtb, Bmat, Cmat);

    k_scan<<<dim3(D_MODEL / 256, B_SZ), blk, 0, stream>>>(x, dtb, Bmat, Cmat,
                                                          A_log, D_par, out);
}
